// KNRM_66829691126374
// MI455X (gfx1250) — compile-verified, matrix operand order checked by reading, not run on hardware
//
#include <hip/hip_runtime.h>
#include <hip/hip_bf16.h>
#include <cmath>

typedef __attribute__((ext_vector_type(16))) _Float16 v16h;
typedef __attribute__((ext_vector_type(8)))  _Float16 v8h;
typedef __attribute__((ext_vector_type(8)))  float    v8f;

#define B_SZ  1024
#define QLEN  64
#define DLEN  256
#define EMBD  128
#define NK    21

// ---------------------------------------------------------------------------
// Main fused kernel: one block = one (batch, pair).
//   Phase 1: gather + L2-normalize embeddings (f32 -> f16 into LDS)
//   Phase 2: WMMA cosine tiles + anchored-chain Gaussian kernel sums
//   Phase 3: exact-match kernel from ids, log1p, reduce over Q -> ko[pair][b][21]
// ---------------------------------------------------------------------------
__global__ __launch_bounds__(256)
void knrm_main(const int* __restrict__ q1, const int* __restrict__ d1,
               const int* __restrict__ q2, const int* __restrict__ d2,
               const float* __restrict__ emb, float* __restrict__ ko)
{
    __shared__ __align__(16) _Float16 sQ[QLEN][EMBD];   // 16 KB
    __shared__ __align__(16) _Float16 sD[DLEN][EMBD];   // 64 KB
    __shared__ int   sQid[QLEN];
    __shared__ int   sDid[DLEN];
    __shared__ float sAcc[QLEN][NK];                    // 5.25 KB

    const int tid  = threadIdx.x;
    const int lane = tid & 31;
    const int wave = tid >> 5;
    const int b    = blockIdx.x;
    const int pair = blockIdx.y;

    const int* qids = pair ? q2 : q1;
    const int* dids = pair ? d2 : d1;

    // zero accumulator table + stage ids
    for (int i = tid; i < QLEN * NK; i += 256) ((float*)sAcc)[i] = 0.0f;
    if (tid < QLEN) sQid[tid] = qids[(size_t)b * QLEN + tid];
    sDid[tid] = dids[(size_t)b * DLEN + tid];
    __syncthreads();

    // ---------------- Phase 1: gather + normalize (8 lanes per vector) -----
    {
        const int sub = lane & 7;                 // 16 floats per lane
        for (int v = wave * 4 + (lane >> 3); v < QLEN + DLEN; v += 32) {
            const int id = (v < QLEN) ? sQid[v] : sDid[v - QLEN];
            const float4* src = (const float4*)(emb + (size_t)id * EMBD) + sub * 4;
            float4 a0 = src[0], a1 = src[1], a2 = src[2], a3 = src[3];
            float ss = a0.x*a0.x + a0.y*a0.y + a0.z*a0.z + a0.w*a0.w
                     + a1.x*a1.x + a1.y*a1.y + a1.z*a1.z + a1.w*a1.w
                     + a2.x*a2.x + a2.y*a2.y + a2.z*a2.z + a2.w*a2.w
                     + a3.x*a3.x + a3.y*a3.y + a3.z*a3.z + a3.w*a3.w;
            ss += __shfl_xor(ss, 1);
            ss += __shfl_xor(ss, 2);
            ss += __shfl_xor(ss, 4);
            const float inv = 1.0f / (sqrtf(ss) + 1e-13f);   // matches reference (pad row -> 0)
            _Float16* dst = (v < QLEN) ? &sQ[v][0] : &sD[v - QLEN][0];
            v8h h0, h1;
            h0[0]=(_Float16)(a0.x*inv); h0[1]=(_Float16)(a0.y*inv);
            h0[2]=(_Float16)(a0.z*inv); h0[3]=(_Float16)(a0.w*inv);
            h0[4]=(_Float16)(a1.x*inv); h0[5]=(_Float16)(a1.y*inv);
            h0[6]=(_Float16)(a1.z*inv); h0[7]=(_Float16)(a1.w*inv);
            h1[0]=(_Float16)(a2.x*inv); h1[1]=(_Float16)(a2.y*inv);
            h1[2]=(_Float16)(a2.z*inv); h1[3]=(_Float16)(a2.w*inv);
            h1[4]=(_Float16)(a3.x*inv); h1[5]=(_Float16)(a3.y*inv);
            h1[6]=(_Float16)(a3.z*inv); h1[7]=(_Float16)(a3.w*inv);
            *(v8h*)(dst + sub * 16)     = h0;
            *(v8h*)(dst + sub * 16 + 8) = h1;
        }
    }
    __syncthreads();

    // ---------------- Phase 2: WMMA tiles + kernel sums --------------------
    const int half  = lane >> 4;
    const int lid   = lane & 15;
    const int qtile = wave & 3;          // 4 q-tiles of 16 rows
    const int khalf = wave >> 2;         // split 20 soft kernels across 2 waves
    const int kbase = khalf * 10;

    // anchored-chain constants (uniform per wave)
    float mu_g[2], e9_g[2];
    #pragma unroll
    for (int g = 0; g < 2; ++g) {
        const int a = kbase + g * 5;
        mu_g[g] = -0.95f + 0.1f * (float)a;
        e9_g[g] = __expf(9.0f - (float)a);        // e^{9-a}
    }
    const float INV_E = 0.36787944117144233f;

    // preload A fragments for this wave's q-tile (K = 128 -> 4 fragments)
    v16h afrag[4];
    #pragma unroll
    for (int kb = 0; kb < 4; ++kb) {
        const _Float16* ap = &sQ[qtile * 16 + lid][kb * 32 + half * 8];
        v8h lo = *(const v8h*)ap;
        v8h hi = *(const v8h*)(ap + 16);
        #pragma unroll
        for (int i = 0; i < 8; ++i) { afrag[kb][i] = lo[i]; afrag[kb][i + 8] = hi[i]; }
    }

    float acc[8][10];
    #pragma unroll
    for (int j = 0; j < 8; ++j)
        #pragma unroll
        for (int k = 0; k < 10; ++k) acc[j][k] = 0.0f;

    for (int dt = 0; dt < 16; ++dt) {
        v8f c = {};
        #pragma unroll
        for (int kb = 0; kb < 4; ++kb) {
            const _Float16* bp = &sD[dt * 16 + lid][kb * 32 + half * 8];
            v8h lo = *(const v8h*)bp;
            v8h hi = *(const v8h*)(bp + 16);
            v16h bf;
            #pragma unroll
            for (int i = 0; i < 8; ++i) { bf[i] = lo[i]; bf[i + 8] = hi[i]; }
            c = __builtin_amdgcn_wmma_f32_16x16x32_f16(
                    false, afrag[kb], false, bf, (short)0, c, false, false);
        }
        #pragma unroll
        for (int j = 0; j < 8; ++j) {
            const float m = c[j];
            const float R = __expf(10.0f * m);         // chain ratio exp(10m)
            #pragma unroll
            for (int g = 0; g < 2; ++g) {
                const float t = m - mu_g[g];
                float v = __expf(-50.0f * t * t);      // anchor Gaussian
                float f = R * e9_g[g];                 // step factor e^{9-k}*e^{10m}
                #pragma unroll
                for (int s = 0; s < 5; ++s) {
                    acc[j][g * 5 + s] += v;
                    v *= f;
                    f *= INV_E;
                }
            }
        }
    }

    // flush per-lane partial sums (row = qtile*16 + j + 8*half per C layout)
    #pragma unroll
    for (int j = 0; j < 8; ++j) {
        const int row = qtile * 16 + j + half * 8;
        #pragma unroll
        for (int k = 0; k < 10; ++k)
            atomicAdd(&sAcc[row][kbase + k], acc[j][k]);
    }
    __syncthreads();

    // ---------------- Phase 3: exact kernel + log1p + Q-reduction ----------
    if (tid < QLEN) {
        const int qid = sQid[tid];
        int cnt = 0;
        if (qid != 0)
            for (int d = 0; d < DLEN; ++d) cnt += (sDid[d] == qid);
        sAcc[tid][NK - 1] = (float)cnt;    // sigma=0.001 kernel == exact id matches
    }
    __syncthreads();

    if (tid < NK) {
        float s = 0.0f;
        for (int r = 0; r < QLEN; ++r) s += log1pf(sAcc[r][tid]);
        ko[((size_t)pair * B_SZ + b) * NK + tid] = s;
    }
}

// ---------------------------------------------------------------------------
// Tiny MLP + pairwise sigmoid: one thread per batch row.
// ---------------------------------------------------------------------------
__global__ __launch_bounds__(256)
void knrm_mlp(const float* __restrict__ ko,
              const float* __restrict__ W1, const float* __restrict__ b1,
              const float* __restrict__ W2, const float* __restrict__ b2,
              const float* __restrict__ W3, const float* __restrict__ b3,
              float* __restrict__ out)
{
    const int b = blockIdx.x * blockDim.x + threadIdx.x;
    if (b >= B_SZ) return;

    float l[2];
    #pragma unroll
    for (int p = 0; p < 2; ++p) {
        const float* x = ko + ((size_t)p * B_SZ + b) * NK;
        float h1[10];
        #pragma unroll
        for (int j = 0; j < 10; ++j) {
            float s = b1[j];
            for (int i = 0; i < NK; ++i) s += fmaxf(x[i], 0.0f) * W1[i * 10 + j];
            h1[j] = s;
        }
        float h2[5];
        #pragma unroll
        for (int j = 0; j < 5; ++j) {
            float s = b2[j];
            #pragma unroll
            for (int i = 0; i < 10; ++i) s += fmaxf(h1[i], 0.0f) * W2[i * 5 + j];
            h2[j] = s;
        }
        float s3 = b3[0];
        #pragma unroll
        for (int i = 0; i < 5; ++i) s3 += fmaxf(h2[i], 0.0f) * W3[i];
        l[p] = s3;
    }
    out[b] = 1.0f / (1.0f + __expf(-(l[0] - l[1])));
}

// ---------------------------------------------------------------------------
extern "C" void kernel_launch(void* const* d_in, const int* in_sizes, int n_in,
                              void* d_out, int out_size, void* d_ws, size_t ws_size,
                              hipStream_t stream)
{
    (void)in_sizes; (void)n_in; (void)out_size; (void)ws_size;
    const int*   q1  = (const int*)d_in[0];
    const int*   d1  = (const int*)d_in[1];
    const int*   q2  = (const int*)d_in[2];
    const int*   d2  = (const int*)d_in[3];
    const float* emb = (const float*)d_in[4];
    const float* W1  = (const float*)d_in[5];
    const float* b1  = (const float*)d_in[6];
    const float* W2  = (const float*)d_in[7];
    const float* b2  = (const float*)d_in[8];
    const float* W3  = (const float*)d_in[9];
    const float* b3  = (const float*)d_in[10];

    float* ko = (float*)d_ws;   // [2][1024][21] f32 = 168 KB

    knrm_main<<<dim3(B_SZ, 2), dim3(256), 0, stream>>>(q1, d1, q2, d2, emb, ko);
    knrm_mlp<<<dim3(B_SZ / 256), dim3(256), 0, stream>>>(ko, W1, b1, W2, b2, W3, b3,
                                                         (float*)d_out);
}
